// MyViTModel_51539607998
// MI455X (gfx1250) — matrix operand order repaired and checked by reading, not compiled, not run on hardware
//
#include <hip/hip_runtime.h>
#include <hip/hip_bf16.h>
#include <math.h>

typedef unsigned short u16;
typedef __attribute__((ext_vector_type(16))) __bf16 v16bf;
typedef __attribute__((ext_vector_type(8)))  float  v8f;

#define BATCH   64
#define SEQ     197
#define SPAD    224          // SEQ padded to multiple of 32 (WMMA K step)
#define HID     768
#define HEADS   12
#define DH      64
#define NBLK    12
#define MLPDIM  3072
#define NPATCH  196
#define OUTC    1000
#define MTOK    (BATCH*SEQ)     // 12608
#define MPATCH  (BATCH*NPATCH)  // 12544

#define BM 256
#define BN 64
#define BK 32

enum { EPI_BF16 = 0, EPI_VT, EPI_SCORES, EPI_RESID, EPI_GELU, EPI_F32B, EPI_EMBED };

__device__ __forceinline__ u16 f2bf(float x) {
  unsigned u = __float_as_uint(x);
  unsigned r = u + 0x7FFFu + ((u >> 16) & 1u);   // round-to-nearest-even
  return (u16)(r >> 16);
}

union Frag {
  uint4  q[2];
  v16bf  v;
};

// ---------------------------------------------------------------------------
// Generic bf16 WMMA GEMM:  C[m,n] (epilogue)  sum_k A[m,k] * W[n,k]
// Both A and W are bf16 row-major with K contiguous.  K must be multiple of 32.
// Double-buffered LDS; for fully in-bounds tiles the next K-tile is DMA'd
// straight into LDS with GLOBAL_LOAD_ASYNC_TO_LDS_B128 (ASYNCcnt) while WMMAs
// consume the current tile; edge tiles fall back to guarded register prefetch.
// 8 waves (wave32) as 4x2; each wave owns 64x32 (8 WMMA tiles per K-step).
// grid.z = Z; z split as (bz = z/zdiv, hz = z%zdiv); per-operand element
// offsets applied as base + bz*OffB + hz*OffH.
// ---------------------------------------------------------------------------
template<int EPI>
__global__ __launch_bounds__(256) void gemm_bf16_wmma(
    const u16* __restrict__ A, int lda, long long aOffB, long long aOffH,
    const u16* __restrict__ W, int ldw, long long wOffB, long long wOffH,
    void* __restrict__ Cv, int ldc, long long cOffB, long long cOffH,
    const float* __restrict__ bias, int biasOffH,
    int M, int N, int K, int zdiv,
    const float* __restrict__ aux,
    int p0, int p1, int p2)
{
  int z  = blockIdx.z;
  int bz = z / zdiv, hz = z % zdiv;
  const u16* Ab = A + bz * aOffB + hz * aOffH;
  const u16* Wb = W + bz * wOffB + hz * wOffH;
  long long  cOff = bz * cOffB + hz * cOffH;

  __shared__ __align__(16) u16 As[2][BM * BK];   // 2 x 16 KB
  __shared__ __align__(16) u16 Bs[2][BN * BK];   // 2 x  4 KB

  int t    = threadIdx.x;
  int lane = t & 31, wave = t >> 5;
  int wm   = wave & 3, wn = wave >> 2;           // 4 x 2 wave grid
  int lrow = lane & 15, lhi = lane >> 4;

  int mBase = blockIdx.y * BM;
  int nBase = blockIdx.x * BN;
  const bool fullM = (mBase + BM) <= M;
  const bool fullN = (nBase + BN) <= N;
  const bool fullTile = fullM && fullN;

  // staging assignments (fixed per thread)
  const int aRow0 = t >> 2;            // + i*64 for i in 0..3
  const int aCol  = (t & 3) * 8;
  const int bRow  = t >> 2;
  const int bCol  = (t & 3) * 8;

  v8f acc[4][2];
  #pragma unroll
  for (int i = 0; i < 4; ++i) { acc[i][0] = {}; acc[i][1] = {}; }

  uint4 pa[4], pb;

  // async DMA of one K-tile straight into LDS (full tiles only)
  auto asy = [&](int kt, int buf) {
    long long kb = (long long)kt * BK;
    #pragma unroll
    for (int i = 0; i < 4; ++i) {
      const u16* g = Ab + (long long)(mBase + aRow0 + i * 64) * lda + kb + aCol;
      unsigned   l = (unsigned)(size_t)&As[buf][(aRow0 + i * 64) * BK + aCol];
      asm volatile("global_load_async_to_lds_b128 %0, %1, off"
                   :: "v"(l), "v"((unsigned long long)(size_t)g) : "memory");
    }
    {
      const u16* g = Wb + (long long)(nBase + bRow) * ldw + kb + bCol;
      unsigned   l = (unsigned)(size_t)&Bs[buf][bRow * BK + bCol];
      asm volatile("global_load_async_to_lds_b128 %0, %1, off"
                   :: "v"(l), "v"((unsigned long long)(size_t)g) : "memory");
    }
  };
  auto g2r = [&](int kt) {
    long long kb = (long long)kt * BK;
    #pragma unroll
    for (int i = 0; i < 4; ++i) {
      int gm = mBase + aRow0 + i * 64;
      pa[i] = make_uint4(0u, 0u, 0u, 0u);
      if (gm < M) pa[i] = *(const uint4*)(Ab + (long long)gm * lda + kb + aCol);
    }
    int gn = nBase + bRow;
    pb = make_uint4(0u, 0u, 0u, 0u);
    if (gn < N) pb = *(const uint4*)(Wb + (long long)gn * ldw + kb + bCol);
  };
  auto r2s = [&](int buf) {
    #pragma unroll
    for (int i = 0; i < 4; ++i)
      *(uint4*)&As[buf][(aRow0 + i * 64) * BK + aCol] = pa[i];
    *(uint4*)&Bs[buf][bRow * BK + bCol] = pb;
  };

  const int nt = K / BK;
  if (fullTile) {
    asy(0, 0);
    asm volatile("s_wait_asynccnt 0x0" ::: "memory");
  } else {
    g2r(0);
    r2s(0);
  }
  __syncthreads();

  for (int kt = 0; kt < nt; ++kt) {
    int  cur  = kt & 1;
    bool more = (kt + 1) < nt;
    if (more) {
      if (fullTile) asy(kt + 1, (kt + 1) & 1);
      else          g2r(kt + 1);
    }

    // fragments per documented gfx1250 layouts
    Frag af[4], bf_[2];
    #pragma unroll
    for (int tm = 0; tm < 4; ++tm) {
      int r = wm * 64 + tm * 16 + lrow;
      // A 16x32: halves 0..7 = K[lhi*8 .. +7], halves 8..15 = K[16+lhi*8 .. +7]
      af[tm].q[0] = *(const uint4*)&As[cur][r * BK + lhi * 8];
      af[tm].q[1] = *(const uint4*)&As[cur][r * BK + 16 + lhi * 8];
    }
    #pragma unroll
    for (int tn = 0; tn < 2; ++tn) {
      int r = wn * 32 + tn * 16 + lrow;
      // B 32x16 (stored transposed, lane = column): halves 0..7 = K[lhi*8 .. +7], halves 8..15 = K[16+lhi*8 .. +7]
      bf_[tn].q[0] = *(const uint4*)&Bs[cur][r * BK + lhi * 8];
      bf_[tn].q[1] = *(const uint4*)&Bs[cur][r * BK + 16 + lhi * 8];
    }
    #pragma unroll
    for (int tm = 0; tm < 4; ++tm)
      #pragma unroll
      for (int tn = 0; tn < 2; ++tn)
        acc[tm][tn] = __builtin_amdgcn_wmma_f32_16x16x32_bf16(
            false, af[tm].v, false, bf_[tn].v, (short)0, acc[tm][tn], false, false);

    if (more) {
      if (fullTile) asm volatile("s_wait_asynccnt 0x0" ::: "memory");
      else          r2s((kt + 1) & 1);
    }
    __syncthreads();
  }

  // --- epilogue: C layout: VGPR r -> M = r + lhi*8, N = lrow within 16x16 tile
  #pragma unroll
  for (int tm = 0; tm < 4; ++tm) {
    #pragma unroll
    for (int tn = 0; tn < 2; ++tn) {
      #pragma unroll
      for (int r = 0; r < 8; ++r) {
        int m = mBase + wm * 64 + tm * 16 + lhi * 8 + r;
        int n = nBase + wn * 32 + tn * 16 + lrow;
        if (m < M && n < N) {
          float v = acc[tm][tn][r];
          if (EPI == EPI_BF16) {
            float bb = bias ? bias[hz * biasOffH + n] : 0.f;
            ((u16*)Cv)[cOff + (long long)m * ldc + n] = f2bf(v + bb);
          } else if (EPI == EPI_VT) {
            // scatter V transposed: vT[((b*H + h)*DH + n)*SPAD + s]
            float bb = bias ? bias[hz * biasOffH + n] : 0.f;
            int bat = m / p0, s = m % p0;             // p0 = SEQ
            ((u16*)Cv)[(((long long)bat * p2 + hz) * N + n) * p1 + s] = f2bf(v + bb);
          } else if (EPI == EPI_SCORES) {
            ((float*)Cv)[cOff + (long long)m * ldc + n] = v;
          } else if (EPI == EPI_RESID) {
            float bb = bias ? bias[hz * biasOffH + n] : 0.f;
            float* X = (float*)Cv;
            long long idx = cOff + (long long)m * ldc + n;
            X[idx] = X[idx] + v + bb;
          } else if (EPI == EPI_GELU) {
            float x = v + bias[hz * biasOffH + n];
            float g = 0.5f * x * (1.f + erff(x * 0.70710678118654752f));
            ((u16*)Cv)[cOff + (long long)m * ldc + n] = f2bf(g);
          } else if (EPI == EPI_F32B) {
            ((float*)Cv)[cOff + (long long)m * ldc + n] = v + bias[n];
          } else { // EPI_EMBED: patch rows -> X[b*SEQ + 1 + p], + bias + posemb
            int bat = m / p0, p = m % p0;             // p0 = NPATCH
            long long row = (long long)bat * SEQ + 1 + p;
            ((float*)Cv)[row * HID + n] = v + bias[n] + aux[(p + 1) * HID + n];
          }
        }
      }
    }
  }
}

// ---------------------------------------------------------------------------
// LayerNorm over 768 columns, one workgroup (256 threads) per row, bf16 out
// ---------------------------------------------------------------------------
__global__ __launch_bounds__(256) void layernorm_768(
    const float* __restrict__ X, const float* __restrict__ g,
    const float* __restrict__ b, u16* __restrict__ out)
{
  __shared__ float red[256];
  long long row = blockIdx.x;
  const float* xr = X + row * HID;
  int tid = threadIdx.x;
  float v0 = xr[tid], v1 = xr[tid + 256], v2 = xr[tid + 512];
  red[tid] = v0 + v1 + v2;
  __syncthreads();
  for (int off = 128; off > 0; off >>= 1) {
    if (tid < off) red[tid] += red[tid + off];
    __syncthreads();
  }
  float mu = red[0] * (1.f / HID);
  __syncthreads();
  float d0 = v0 - mu, d1 = v1 - mu, d2 = v2 - mu;
  red[tid] = d0 * d0 + d1 * d1 + d2 * d2;
  __syncthreads();
  for (int off = 128; off > 0; off >>= 1) {
    if (tid < off) red[tid] += red[tid + off];
    __syncthreads();
  }
  float rstd = rsqrtf(red[0] * (1.f / HID) + 1e-5f);
  u16* orow = out + row * HID;
  orow[tid]       = f2bf(d0 * rstd * g[tid]       + b[tid]);
  orow[tid + 256] = f2bf(d1 * rstd * g[tid + 256] + b[tid + 256]);
  orow[tid + 512] = f2bf(d2 * rstd * g[tid + 512] + b[tid + 512]);
}

// ---------------------------------------------------------------------------
// Attention softmax: one WG per (b,h,s) row; scale 1/sqrt(64); bf16 out with
// zero padding columns [SEQ, SPAD)
// ---------------------------------------------------------------------------
__global__ __launch_bounds__(256) void attn_softmax(
    const float* __restrict__ scores, u16* __restrict__ attn)
{
  __shared__ float red[256];
  long long row = blockIdx.x;
  const float* sr = scores + row * SPAD;
  u16* ar = attn + row * SPAD;
  int tid = threadIdx.x;
  float v = (tid < SEQ) ? sr[tid] * 0.125f : -3.0e38f;
  red[tid] = v;
  __syncthreads();
  for (int off = 128; off > 0; off >>= 1) {
    if (tid < off) red[tid] = fmaxf(red[tid], red[tid + off]);
    __syncthreads();
  }
  float mx = red[0];
  __syncthreads();
  float e = (tid < SEQ) ? __expf(v - mx) : 0.f;
  red[tid] = e;
  __syncthreads();
  for (int off = 128; off > 0; off >>= 1) {
    if (tid < off) red[tid] += red[tid + off];
    __syncthreads();
  }
  float inv = 1.f / red[0];
  if (tid < SPAD) ar[tid] = f2bf((tid < SEQ) ? e * inv : 0.f);
}

// ---------------------------------------------------------------------------
// Final softmax over 1000 classes, one WG per batch row, f32 out
// ---------------------------------------------------------------------------
__global__ __launch_bounds__(256) void out_softmax(
    const float* __restrict__ logits, float* __restrict__ out)
{
  __shared__ float red[256];
  int row = blockIdx.x;
  const float* lr = logits + (long long)row * OUTC;
  int tid = threadIdx.x;
  float v[4]; float mx = -3.0e38f;
  #pragma unroll
  for (int i = 0; i < 4; ++i) {
    int c = tid + i * 256;
    v[i] = (c < OUTC) ? lr[c] : -3.0e38f;
    mx = fmaxf(mx, v[i]);
  }
  red[tid] = mx;
  __syncthreads();
  for (int off = 128; off > 0; off >>= 1) {
    if (tid < off) red[tid] = fmaxf(red[tid], red[tid + off]);
    __syncthreads();
  }
  float M = red[0];
  __syncthreads();
  float e[4]; float s = 0.f;
  #pragma unroll
  for (int i = 0; i < 4; ++i) {
    int c = tid + i * 256;
    e[i] = (c < OUTC) ? __expf(v[i] - M) : 0.f;
    s += e[i];
  }
  red[tid] = s;
  __syncthreads();
  for (int off = 128; off > 0; off >>= 1) {
    if (tid < off) red[tid] += red[tid + off];
    __syncthreads();
  }
  float inv = 1.f / red[0];
  float* orow = out + (long long)row * OUTC;
  #pragma unroll
  for (int i = 0; i < 4; ++i) {
    int c = tid + i * 256;
    if (c < OUTC) orow[c] = e[i] * inv;
  }
}

// ---------------------------------------------------------------------------
// Elementwise helpers
// ---------------------------------------------------------------------------
__global__ void k_f32_to_bf16(const float* __restrict__ in, u16* __restrict__ out,
                              long long n) {
  long long i = (long long)blockIdx.x * 256 + threadIdx.x;
  if (i < n) out[i] = f2bf(in[i]);
}

__global__ void k_zero_u16(u16* __restrict__ p, long long n) {
  long long i = (long long)blockIdx.x * 256 + threadIdx.x;
  if (i < n) p[i] = 0;
}

__global__ void k_posemb(float* __restrict__ pos) {
  int idx = blockIdx.x * 256 + threadIdx.x;
  if (idx >= SEQ * HID) return;
  int i = idx / HID, j = idx % HID;
  int even = ((j & 1) == 0);
  float expo = (float)(even ? j : (j - 1)) * (1.f / HID);
  float ang = (float)i / powf(10000.f, expo);
  pos[idx] = even ? sinf(ang) : cosf(ang);
}

__global__ void k_patchify(const float* __restrict__ img, u16* __restrict__ out) {
  long long idx = (long long)blockIdx.x * 256 + threadIdx.x;
  if (idx >= (long long)MPATCH * HID) return;
  int n = (int)(idx % HID);
  long long m = idx / HID;
  int b = (int)(m / NPATCH), p = (int)(m % NPATCH);
  int ry = p / 14, rx = p % 14;
  int c = n / 256, rem = n % 256;
  int py = rem / 16, px = rem % 16;
  long long src = (((long long)b * 3 + c) * 224 + (ry * 16 + py)) * 224 + (rx * 16 + px);
  out[idx] = f2bf(img[src]);
}

__global__ void k_cls_init(float* __restrict__ X, const float* __restrict__ cls,
                           const float* __restrict__ pos) {
  int idx = blockIdx.x * 256 + threadIdx.x;
  if (idx >= BATCH * HID) return;
  int b = idx / HID, n = idx % HID;
  X[(long long)b * SEQ * HID + n] = cls[n] + pos[n];
}

__global__ void k_cls_extract(const float* __restrict__ X, u16* __restrict__ out) {
  int idx = blockIdx.x * 256 + threadIdx.x;
  if (idx >= BATCH * HID) return;
  int b = idx / HID, n = idx % HID;
  out[idx] = f2bf(X[(long long)b * SEQ * HID + n]);
}

// ---------------------------------------------------------------------------
// Host-side GEMM dispatcher
// ---------------------------------------------------------------------------
#define GEMM_ARGS pA, lda, aOffB, aOffH, pW, ldw, wOffB, wOffH, pC, ldc, cOffB, cOffH, \
                  pBias, biasOffH, M, N, K, zdiv, pAux, p0, p1, p2
static void run_gemm(hipStream_t st, int epi,
                     const u16* pA, int lda, long long aOffB, long long aOffH,
                     const u16* pW, int ldw, long long wOffB, long long wOffH,
                     void* pC, int ldc, long long cOffB, long long cOffH,
                     const float* pBias, int biasOffH,
                     int M, int N, int K, int Z, int zdiv,
                     const float* pAux, int p0, int p1, int p2)
{
  dim3 grid((N + BN - 1) / BN, (M + BM - 1) / BM, Z);
  dim3 blk(256);
  switch (epi) {
    case EPI_BF16:   gemm_bf16_wmma<EPI_BF16><<<grid, blk, 0, st>>>(GEMM_ARGS);   break;
    case EPI_VT:     gemm_bf16_wmma<EPI_VT><<<grid, blk, 0, st>>>(GEMM_ARGS);     break;
    case EPI_SCORES: gemm_bf16_wmma<EPI_SCORES><<<grid, blk, 0, st>>>(GEMM_ARGS); break;
    case EPI_RESID:  gemm_bf16_wmma<EPI_RESID><<<grid, blk, 0, st>>>(GEMM_ARGS);  break;
    case EPI_GELU:   gemm_bf16_wmma<EPI_GELU><<<grid, blk, 0, st>>>(GEMM_ARGS);   break;
    case EPI_F32B:   gemm_bf16_wmma<EPI_F32B><<<grid, blk, 0, st>>>(GEMM_ARGS);   break;
    default:         gemm_bf16_wmma<EPI_EMBED><<<grid, blk, 0, st>>>(GEMM_ARGS);  break;
  }
}

extern "C" void kernel_launch(void* const* d_in, const int* in_sizes, int n_in,
                              void* d_out, int out_size, void* d_ws, size_t ws_size,
                              hipStream_t stream) {
  (void)in_sizes; (void)n_in; (void)out_size; (void)ws_size;
  const float* images   = (const float*)d_in[0];
  const float* mapper_w = (const float*)d_in[1];
  const float* mapper_b = (const float*)d_in[2];
  const float* cls_tok  = (const float*)d_in[3];
  const float* ln1_g    = (const float*)d_in[4];
  const float* ln1_b    = (const float*)d_in[5];
  const float* qw       = (const float*)d_in[6];
  const float* qb       = (const float*)d_in[7];
  const float* kw       = (const float*)d_in[8];
  const float* kb       = (const float*)d_in[9];
  const float* vw       = (const float*)d_in[10];
  const float* vb       = (const float*)d_in[11];
  const float* ln2_g    = (const float*)d_in[12];
  const float* ln2_b    = (const float*)d_in[13];
  const float* w1       = (const float*)d_in[14];
  const float* b1       = (const float*)d_in[15];
  const float* w2       = (const float*)d_in[16];
  const float* b2       = (const float*)d_in[17];
  const float* head_w   = (const float*)d_in[18];
  const float* head_b   = (const float*)d_in[19];

  char* ws = (char*)d_ws;
  size_t off = 0;
  auto alloc = [&](size_t bytes) -> char* {
    char* p = ws + off;
    off = (off + bytes + 255) & ~(size_t)255;
    return p;
  };

  const long long nQKVw = (long long)NBLK * HEADS * DH * DH;
  u16*   Wm   = (u16*)alloc(2LL * HID * HID);
  u16*   Wq   = (u16*)alloc(2LL * nQKVw);
  u16*   Wk   = (u16*)alloc(2LL * nQKVw);
  u16*   Wv   = (u16*)alloc(2LL * nQKVw);
  u16*   W1   = (u16*)alloc(2LL * NBLK * MLPDIM * HID);
  u16*   W2   = (u16*)alloc(2LL * NBLK * HID * MLPDIM);
  u16*   Wh   = (u16*)alloc(2LL * OUTC * HID);
  u16*   Pbf  = (u16*)alloc(2LL * MPATCH * HID);
  float* X    = (float*)alloc(4LL * MTOK * HID);
  u16*   Hbf  = (u16*)alloc(2LL * MTOK * HID);
  u16*   Qbf  = (u16*)alloc(2LL * MTOK * HID);
  u16*   Kbf  = (u16*)alloc(2LL * MTOK * HID);
  u16*   vT   = (u16*)alloc(2LL * BATCH * HEADS * DH * SPAD);
  float* Sc   = (float*)alloc(4LL * BATCH * HEADS * SEQ * SPAD);
  u16*   At   = (u16*)alloc(2LL * BATCH * HEADS * SEQ * SPAD);
  u16*   Mbf  = (u16*)alloc(2LL * MTOK * MLPDIM);
  u16*   Cls  = (u16*)alloc(2LL * BATCH * HID);
  float* Lg   = (float*)alloc(4LL * BATCH * OUTC);
  float* Pos  = (float*)alloc(4LL * SEQ * HID);

  auto conv = [&](const float* s, u16* d, long long n) {
    k_f32_to_bf16<<<(unsigned)((n + 255) / 256), 256, 0, stream>>>(s, d, n);
  };
  conv(mapper_w, Wm, (long long)HID * HID);
  conv(qw, Wq, nQKVw);
  conv(kw, Wk, nQKVw);
  conv(vw, Wv, nQKVw);
  conv(w1, W1, (long long)NBLK * MLPDIM * HID);
  conv(w2, W2, (long long)NBLK * HID * MLPDIM);
  conv(head_w, Wh, (long long)OUTC * HID);

  k_posemb<<<(SEQ * HID + 255) / 256, 256, 0, stream>>>(Pos);
  {
    long long n = (long long)MPATCH * HID;
    k_patchify<<<(unsigned)((n + 255) / 256), 256, 0, stream>>>(images, Pbf);
  }
  {
    long long n = (long long)BATCH * HEADS * DH * SPAD;
    k_zero_u16<<<(unsigned)((n + 255) / 256), 256, 0, stream>>>(vT, n);
  }
  k_cls_init<<<(BATCH * HID + 255) / 256, 256, 0, stream>>>(X, cls_tok, Pos);

  // Patch embedding: X[b, 1+p, :] = Pbf @ Wm^T + mapper_b + pos[1+p]
  run_gemm(stream, EPI_EMBED, Pbf, HID, 0, 0, Wm, HID, 0, 0,
           X, HID, 0, 0, mapper_b, 0, MPATCH, HID, HID, 1, 1, Pos, NPATCH, 0, 0);

  const long long SH = (long long)SEQ * HID;
  const long long SS = (long long)SEQ * SPAD;

  for (int blk = 0; blk < NBLK; ++blk) {
    // LN1
    layernorm_768<<<MTOK, 256, 0, stream>>>(X, ln1_g + blk * HID, ln1_b + blk * HID, Hbf);
    // Q, K projections (per-head 64x64), bf16 out with bias
    run_gemm(stream, EPI_BF16, Hbf, HID, 0, DH,
             Wq + (long long)blk * HEADS * DH * DH, DH, 0, DH * DH,
             Qbf, HID, 0, DH, qb + blk * HEADS * DH, DH,
             MTOK, DH, DH, HEADS, HEADS, nullptr, 0, 0, 0);
    run_gemm(stream, EPI_BF16, Hbf, HID, 0, DH,
             Wk + (long long)blk * HEADS * DH * DH, DH, 0, DH * DH,
             Kbf, HID, 0, DH, kb + blk * HEADS * DH, DH,
             MTOK, DH, DH, HEADS, HEADS, nullptr, 0, 0, 0);
    // V projection, stored transposed per (b,h): vT[b,h,e,s]
    run_gemm(stream, EPI_VT, Hbf, HID, 0, DH,
             Wv + (long long)blk * HEADS * DH * DH, DH, 0, DH * DH,
             vT, 0, 0, 0, vb + blk * HEADS * DH, DH,
             MTOK, DH, DH, HEADS, HEADS, nullptr, SEQ, SPAD, HEADS);
    // scores[b,h,s,t] = q . k
    run_gemm(stream, EPI_SCORES, Qbf, HID, SH, DH,
             Kbf, HID, SH, DH,
             Sc, SPAD, (long long)HEADS * SS, SS, nullptr, 0,
             SEQ, SEQ, DH, BATCH * HEADS, HEADS, nullptr, 0, 0, 0);
    // softmax -> bf16 attn (padded cols zero)
    attn_softmax<<<BATCH * HEADS * SEQ, 256, 0, stream>>>(Sc, At);
    // out = attn @ v, fused residual add into X
    run_gemm(stream, EPI_RESID, At, SPAD, (long long)HEADS * SS, SS,
             vT, SPAD, (long long)HEADS * DH * SPAD, (long long)DH * SPAD,
             X, HID, SH, DH, nullptr, 0,
             SEQ, DH, SPAD, BATCH * HEADS, HEADS, nullptr, 0, 0, 0);
    // LN2
    layernorm_768<<<MTOK, 256, 0, stream>>>(X, ln2_g + blk * HID, ln2_b + blk * HID, Hbf);
    // MLP1 with fused exact GELU
    run_gemm(stream, EPI_GELU, Hbf, HID, 0, 0,
             W1 + (long long)blk * MLPDIM * HID, HID, 0, 0,
             Mbf, MLPDIM, 0, 0, b1 + blk * MLPDIM, 0,
             MTOK, MLPDIM, HID, 1, 1, nullptr, 0, 0, 0);
    // MLP2 with fused bias + residual add into X
    run_gemm(stream, EPI_RESID, Mbf, MLPDIM, 0, 0,
             W2 + (long long)blk * HID * MLPDIM, MLPDIM, 0, 0,
             X, HID, 0, 0, b2 + blk * HID, 0,
             MTOK, HID, MLPDIM, 1, 1, nullptr, 0, 0, 0);
  }

  // classifier head on CLS rows
  k_cls_extract<<<(BATCH * HID + 255) / 256, 256, 0, stream>>>(X, Cls);
  run_gemm(stream, EPI_F32B, Cls, HID, 0, 0, Wh, HID, 0, 0,
           Lg, OUTC, 0, 0, head_b, 0, BATCH, OUTC, HID, 1, 1, nullptr, 0, 0, 0);
  out_softmax<<<BATCH, 256, 0, stream>>>(Lg, (float*)d_out);
}
